// HCF_7584912245521
// MI455X (gfx1250) — matrix operand order repaired and checked by reading, not compiled, not run on hardware
//
#include <hip/hip_runtime.h>
#include <hip/hip_bf16.h>
#include <math.h>

// ---------------------------------------------------------------------------
// Problem constants
// ---------------------------------------------------------------------------
#define N_USERS  20000
#define N_ITEMS  15000
#define N_TAGS   3000
#define N_GLOBAL (N_USERS + N_ITEMS + N_TAGS)   // 38000
#define DIM      768

#define LDS_STRIDE 40   // padded row stride (ushorts): 80B -> conflict-free frags

typedef __attribute__((ext_vector_type(8)))  float          v8f;
typedef __attribute__((ext_vector_type(16))) __bf16         v16bf;
typedef __attribute__((ext_vector_type(8)))  unsigned short ushort8;

union BFrag {
    v16bf   v;
    ushort8 h[2];
};

__device__ __forceinline__ unsigned short f2bf(float f) {
    unsigned int u = __float_as_uint(f);
    unsigned int r = u + 0x7fffu + ((u >> 16) & 1u);   // round-to-nearest-even
    return (unsigned short)(r >> 16);
}

__device__ __forceinline__ unsigned int pack2bf(float lo, float hi) {
    return (unsigned int)f2bf(lo) | ((unsigned int)f2bf(hi) << 16);
}

// ---------------------------------------------------------------------------
// SpMM: out[row] += val * x[col]  (COO scatter, wave-per-edge, HBM-bound)
// out must be pre-zeroed. Uses native f32 global atomics + prefetch.
// ---------------------------------------------------------------------------
__global__ __launch_bounds__(256) void spmm_kernel(
    const int* __restrict__ rows, const int* __restrict__ cols,
    const float* __restrict__ vals, const float* __restrict__ x,
    float* __restrict__ out, int nnz)
{
    const int gw   = (blockIdx.x * blockDim.x + threadIdx.x) >> 5;
    const int lane = threadIdx.x & 31;
    const int nw   = (gridDim.x * blockDim.x) >> 5;

    for (int e = gw; e < nnz; e += nw) {
        const int   r = rows[e];
        const int   c = cols[e];
        const float v = vals[e];
        if (e + nw < nnz) {   // prefetch next source row -> global_prefetch_b8
            const int cn = cols[e + nw];
            __builtin_prefetch(x + (size_t)cn * DIM, 0, 1);
        }
        const float4* __restrict__ xp = (const float4*)(x + (size_t)c * DIM);
        float*        __restrict__ op = out + (size_t)r * DIM;
#pragma unroll
        for (int d = 0; d < 6; ++d) {             // 6 * 32 lanes * float4 = 768
            const int    i  = lane + d * 32;
            const float4 xv = xp[i];
            const int    b  = i * 4;
            unsafeAtomicAdd(op + b + 0, xv.x * v);
            unsafeAtomicAdd(op + b + 1, xv.y * v);
            unsafeAtomicAdd(op + b + 2, xv.z * v);
            unsafeAtomicAdd(op + b + 3, xv.w * v);
        }
    }
}

// ---------------------------------------------------------------------------
// out = sm(w)[0]*e0 + sm(w)[1]*e1 + sm(w)[2]*e2   (float4, n % 4 == 0)
// ---------------------------------------------------------------------------
__global__ __launch_bounds__(256) void combine3_kernel(
    const float* __restrict__ e0, const float* __restrict__ e1,
    const float* __restrict__ e2, const float* __restrict__ w,
    float* __restrict__ out, long n)
{
    const float w0 = w[0], w1 = w[1], w2 = w[2];
    const float m  = fmaxf(w0, fmaxf(w1, w2));
    float a0 = __expf(w0 - m), a1 = __expf(w1 - m), a2 = __expf(w2 - m);
    const float inv = 1.f / (a0 + a1 + a2);
    a0 *= inv; a1 *= inv; a2 *= inv;
    const float4* p0 = (const float4*)e0;
    const float4* p1 = (const float4*)e1;
    const float4* p2 = (const float4*)e2;
    float4*       po = (float4*)out;
    const long n4 = n >> 2;
    const long stride = (long)gridDim.x * blockDim.x;
    for (long i = (long)blockIdx.x * blockDim.x + threadIdx.x; i < n4; i += stride) {
        const float4 x0 = p0[i], x1 = p1[i], x2 = p2[i];
        float4 r;
        r.x = a0 * x0.x + a1 * x1.x + a2 * x2.x;
        r.y = a0 * x0.y + a1 * x1.y + a2 * x2.y;
        r.z = a0 * x0.z + a1 * x1.z + a2 * x2.z;
        r.w = a0 * x0.w + a1 * x1.w + a2 * x2.w;
        po[i] = r;
    }
}

// out = sm(w)[0]*a + sm(w)[1]*b   (float4, n % 4 == 0)
__global__ __launch_bounds__(256) void combine2_kernel(
    const float* __restrict__ a, const float* __restrict__ b,
    const float* __restrict__ w, float* __restrict__ out, long n)
{
    const float w0 = w[0], w1 = w[1];
    const float m  = fmaxf(w0, w1);
    float a0 = __expf(w0 - m), a1 = __expf(w1 - m);
    const float inv = 1.f / (a0 + a1);
    a0 *= inv; a1 *= inv;
    const float4* pa = (const float4*)a;
    const float4* pb = (const float4*)b;
    float4*       po = (float4*)out;
    const long n4 = n >> 2;
    const long stride = (long)gridDim.x * blockDim.x;
    for (long i = (long)blockIdx.x * blockDim.x + threadIdx.x; i < n4; i += stride) {
        const float4 xa = pa[i], xb = pb[i];
        float4 r;
        r.x = a0 * xa.x + a1 * xb.x;
        r.y = a0 * xa.y + a1 * xb.y;
        r.z = a0 * xa.z + a1 * xb.z;
        r.w = a0 * xa.w + a1 * xb.w;
        po[i] = r;
    }
}

// ---------------------------------------------------------------------------
// GEMM + bias: C[M,N] = A[M,K] @ W[K,N] + bias, fp32 in/out, bf16 WMMA inside.
// Block = 256 threads = 8 waves; block tile BM=128 x BN=64, BK=32.
// Waves: 4 (M) x 2 (N); each wave owns a 32x32 strip -> 2 A-frags x 2 B-frags
// -> 4 x V_WMMA_F32_16X16X32_BF16 per K-step (WMMA:ds_load ratio 1:2).
// Staging is branchless: out-of-range rows/cols are CLAMPED (their products
// land only in masked-out output rows/cols; exact since offsets & N are mult 4).
// LDS rows padded to 40 ushorts (80B) -> conflict-free ds_load_b128 frags.
// Fragment layouts per CDNA5 ISA (16-bit A 16x32, B 32x16, f32 C 16x16).
// Requires K % 32 == 0 (K = 768 here).
// ---------------------------------------------------------------------------
__global__ __launch_bounds__(256) void gemm_bias_kernel(
    const float* __restrict__ A, const float* __restrict__ W,
    const float* __restrict__ bias, float* __restrict__ C,
    int M, int N, int K)
{
    __shared__ __align__(16) unsigned short sA[128 * LDS_STRIDE];  // [m][k]
    __shared__ __align__(16) unsigned short sB[64  * LDS_STRIDE];  // [n][k] (W^T)

    const int tid   = threadIdx.x;
    const int wave  = tid >> 5;
    const int lane  = tid & 31;
    const int waveM = wave >> 1;          // 0..3  (32-row strip)
    const int waveN = wave & 1;           // 0..1  (32-col strip)
    const int g     = lane >> 4;          // lane group
    const int lr    = lane & 15;

    const int bm = blockIdx.y * 128;
    const int bn = blockIdx.x * 64;

    // cooperative-load coordinates: A tile 128 x 32 (16 k's per thread)
    const int am  = tid >> 1;             // 0..127  A row within tile
    const int ak  = (tid & 1) * 16;       // 0 or 16
    const int arow_g = min(bm + am, M - 1);
    const float4* __restrict__ apv =
        (const float4*)(A + (size_t)arow_g * K + ak);

    // B tile 32(k) x 64(n)
    const int kk  = (tid & 15) * 2;       // 0..30   K pair within slab
    const int nn0 = (tid >> 4) * 4;       // 0..60   4 consecutive cols
    const int c0  = min(bn + nn0, N - 4); // exact clamp (bn+nn0 = 0 mod 4, N = 0 mod 4)

    v8f acc00 = {};   // (a0,b0)
    v8f acc01 = {};   // (a0,b1)
    v8f acc10 = {};   // (a1,b0)
    v8f acc11 = {};   // (a1,b1)

    for (int k0 = 0; k0 < K; k0 += 32) {
        // ---- stage A slab: 4x global b128 -> pack -> 2x ds_store_b128 ----
        {
            const float4 f0 = apv[(k0 >> 2) + 0];
            const float4 f1 = apv[(k0 >> 2) + 1];
            const float4 f2 = apv[(k0 >> 2) + 2];
            const float4 f3 = apv[(k0 >> 2) + 3];
            ushort8 p0, p1;
            p0[0] = f2bf(f0.x); p0[1] = f2bf(f0.y);
            p0[2] = f2bf(f0.z); p0[3] = f2bf(f0.w);
            p0[4] = f2bf(f1.x); p0[5] = f2bf(f1.y);
            p0[6] = f2bf(f1.z); p0[7] = f2bf(f1.w);
            p1[0] = f2bf(f2.x); p1[1] = f2bf(f2.y);
            p1[2] = f2bf(f2.z); p1[3] = f2bf(f2.w);
            p1[4] = f2bf(f3.x); p1[5] = f2bf(f3.y);
            p1[6] = f2bf(f3.z); p1[7] = f2bf(f3.w);
            *(ushort8*)&sA[am * LDS_STRIDE + ak]     = p0;
            *(ushort8*)&sA[am * LDS_STRIDE + ak + 8] = p1;
        }
        // ---- stage W slab transposed: 2x global b128 -> 4x ds_store_b32 ----
        {
            const float4 r0 = *(const float4*)(W + (size_t)(k0 + kk)     * N + c0);
            const float4 r1 = *(const float4*)(W + (size_t)(k0 + kk + 1) * N + c0);
            *(unsigned int*)&sB[(nn0 + 0) * LDS_STRIDE + kk] = pack2bf(r0.x, r1.x);
            *(unsigned int*)&sB[(nn0 + 1) * LDS_STRIDE + kk] = pack2bf(r0.y, r1.y);
            *(unsigned int*)&sB[(nn0 + 2) * LDS_STRIDE + kk] = pack2bf(r0.z, r1.z);
            *(unsigned int*)&sB[(nn0 + 3) * LDS_STRIDE + kk] = pack2bf(r0.w, r1.w);
        }
        __syncthreads();

        // ---- fragments: A 16x32 (two 16B chunks), B 32x16 (two 16B chunks) ----
        BFrag a0, a1, b0, b1;
        const int arow0 = (waveM * 32 + lr)      * LDS_STRIDE;
        const int arow1 = (waveM * 32 + 16 + lr) * LDS_STRIDE;
        a0.h[0] = *(const ushort8*)&sA[arow0 + 8 * g];
        a0.h[1] = *(const ushort8*)&sA[arow0 + 16 + 8 * g];
        a1.h[0] = *(const ushort8*)&sA[arow1 + 8 * g];
        a1.h[1] = *(const ushort8*)&sA[arow1 + 16 + 8 * g];
        const int brow0 = (waveN * 32 + lr)      * LDS_STRIDE + 8 * g;
        const int brow1 = (waveN * 32 + 16 + lr) * LDS_STRIDE + 8 * g;
        b0.h[0] = *(const ushort8*)&sB[brow0];
        b0.h[1] = *(const ushort8*)&sB[brow0 + 16];
        b1.h[0] = *(const ushort8*)&sB[brow1];
        b1.h[1] = *(const ushort8*)&sB[brow1 + 16];

        acc00 = __builtin_amdgcn_wmma_f32_16x16x32_bf16(
            false, a0.v, false, b0.v, (short)0, acc00, false, false);
        acc01 = __builtin_amdgcn_wmma_f32_16x16x32_bf16(
            false, a0.v, false, b1.v, (short)0, acc01, false, false);
        acc10 = __builtin_amdgcn_wmma_f32_16x16x32_bf16(
            false, a1.v, false, b0.v, (short)0, acc10, false, false);
        acc11 = __builtin_amdgcn_wmma_f32_16x16x32_bf16(
            false, a1.v, false, b1.v, (short)0, acc11, false, false);

        __syncthreads();
    }

    const int col0 = bn + waveN * 32 + lr;
    const int col1 = col0 + 16;
    const float bv0 = (col0 < N) ? bias[col0] : 0.f;
    const float bv1 = (col1 < N) ? bias[col1] : 0.f;
#pragma unroll
    for (int r = 0; r < 8; ++r) {
        const int row0 = bm + waveM * 32 + g * 8 + r;
        const int row1 = row0 + 16;
        if (row0 < M) {
            if (col0 < N) C[(size_t)row0 * N + col0] = acc00[r] + bv0;
            if (col1 < N) C[(size_t)row0 * N + col1] = acc01[r] + bv1;
        }
        if (row1 < M) {
            if (col0 < N) C[(size_t)row1 * N + col0] = acc10[r] + bv0;
            if (col1 < N) C[(size_t)row1 * N + col1] = acc11[r] + bv1;
        }
    }
}

// ---------------------------------------------------------------------------
// Fused LayerNorm (D=768) + exact GELU, in place. One block per row.
// ---------------------------------------------------------------------------
__global__ __launch_bounds__(256) void ln_gelu_kernel(
    float* __restrict__ h, const float* __restrict__ gam,
    const float* __restrict__ bet)
{
    __shared__ float red[256];
    __shared__ float stats[2];
    float* __restrict__ p = h + (size_t)blockIdx.x * DIM;
    const int tid = threadIdx.x;

    const float x0 = p[tid], x1 = p[tid + 256], x2 = p[tid + 512];

    red[tid] = x0 + x1 + x2;
    __syncthreads();
    for (int s = 128; s > 0; s >>= 1) {
        if (tid < s) red[tid] += red[tid + s];
        __syncthreads();
    }
    if (tid == 0) stats[0] = red[0] * (1.0f / DIM);
    __syncthreads();
    const float mu = stats[0];

    const float d0 = x0 - mu, d1 = x1 - mu, d2 = x2 - mu;
    red[tid] = d0 * d0 + d1 * d1 + d2 * d2;
    __syncthreads();
    for (int s = 128; s > 0; s >>= 1) {
        if (tid < s) red[tid] += red[tid + s];
        __syncthreads();
    }
    if (tid == 0) stats[1] = rsqrtf(red[0] * (1.0f / DIM) + 1e-5f);
    __syncthreads();
    const float rstd = stats[1];

#pragma unroll
    for (int j = 0; j < 3; ++j) {
        const int   idx = tid + j * 256;
        const float d   = (j == 0 ? d0 : (j == 1 ? d1 : d2));
        float y = d * rstd * gam[idx] + bet[idx];
        y = 0.5f * y * (1.0f + erff(y * 0.70710678118654752f));   // exact GELU
        p[idx] = y;
    }
}

// ---------------------------------------------------------------------------
// Host-side orchestration
// ---------------------------------------------------------------------------
static void run_spmm(const int* idx, const float* val, int nnz,
                     const float* x, float* out, int nrows, hipStream_t stream)
{
    hipMemsetAsync(out, 0, (size_t)nrows * DIM * sizeof(float), stream);
    int blocks = (nnz + 7) / 8;
    if (blocks > 8192) blocks = 8192;
    spmm_kernel<<<blocks, 256, 0, stream>>>(idx, idx + nnz, val, x, out, nnz);
}

// propagate with the redundant 3rd layer elided: out = sm(w)·[e0, L1, L2]
static void run_propagate(const int* idx1, const float* val1, int nnz1,
                          const int* idx2, const float* val2, int nnz2,
                          const float* e0, const float* w,
                          int n1, int n2,
                          float* T, float* L1, float* L2, float* out,
                          hipStream_t stream)
{
    run_spmm(idx2, val2, nnz2, e0, T,  n2, stream);
    run_spmm(idx1, val1, nnz1, T,  L1, n1, stream);
    run_spmm(idx2, val2, nnz2, L1, T,  n2, stream);
    run_spmm(idx1, val1, nnz1, T,  L2, n1, stream);
    const long n = (long)n1 * DIM;
    combine3_kernel<<<2048, 256, 0, stream>>>(e0, L1, L2, w, out, n);
}

static void run_gemm(const float* A, const float* W, const float* b, float* C,
                     int M, int N, int K, hipStream_t stream)
{
    dim3 grid((N + 63) / 64, (M + 127) / 128);
    gemm_bias_kernel<<<grid, 256, 0, stream>>>(A, W, b, C, M, N, K);
}

// proj params flattened in pytree (sorted-key) order: b1,b2,be,g,w1,w2
static void run_proj(const float* x, void* const* p, float* H, float* out,
                     int M, hipStream_t stream)
{
    const float* b1 = (const float*)p[0];
    const float* b2 = (const float*)p[1];
    const float* be = (const float*)p[2];
    const float* g  = (const float*)p[3];
    const float* w1 = (const float*)p[4];
    const float* w2 = (const float*)p[5];
    run_gemm(x, w1, b1, H, M, DIM, DIM, stream);
    ln_gelu_kernel<<<M, 256, 0, stream>>>(H, g, be);
    run_gemm(H, w2, b2, out, M, DIM, DIM, stream);
}

extern "C" void kernel_launch(void* const* d_in, const int* in_sizes, int n_in,
                              void* d_out, int out_size, void* d_ws, size_t ws_size,
                              hipStream_t stream)
{
    // ---- inputs ----
    const int*   a_mc1_i = (const int*)d_in[0];   const float* a_mc1_v = (const float*)d_in[1];
    const int*   a_mc2_i = (const int*)d_in[2];   const float* a_mc2_v = (const float*)d_in[3];
    const int*   a_ac1_i = (const int*)d_in[4];   const float* a_ac1_v = (const float*)d_in[5];
    const int*   a_ac2_i = (const int*)d_in[6];   const float* a_ac2_v = (const float*)d_in[7];
    const int*   a_mt1_i = (const int*)d_in[8];   const float* a_mt1_v = (const float*)d_in[9];
    const int*   a_mt2_i = (const int*)d_in[10];  const float* a_mt2_v = (const float*)d_in[11];
    const int*   a_at1_i = (const int*)d_in[12];  const float* a_at1_v = (const float*)d_in[13];
    const int*   a_at2_i = (const int*)d_in[14];  const float* a_at2_v = (const float*)d_in[15];
    const int*   g1_i    = (const int*)d_in[16];  const float* g1_v    = (const float*)d_in[17];
    const int*   g2_i    = (const int*)d_in[18];  const float* g2_v    = (const float*)d_in[19];
    const float* emb_mc  = (const float*)d_in[20];
    const float* emb_ac  = (const float*)d_in[21];
    const float* emb_mt  = (const float*)d_in[22];
    const float* emb_at  = (const float*)d_in[23];
    const float* emb_g   = (const float*)d_in[24];
    const float* u_w     = (const float*)d_in[25];
    const float* i_w     = (const float*)d_in[26];
    const float* mt_w    = (const float*)d_in[27];
    const float* at_w    = (const float*)d_in[28];
    const float* g_w     = (const float*)d_in[29];
    const float* mvw     = (const float*)d_in[30];
    const float* avw     = (const float*)d_in[31];
    const float* mlg     = (const float*)d_in[32];
    const float* alg     = (const float*)d_in[33];
    void* const* p_mloc  = d_in + 34;   // b1,b2,be,g,w1,w2
    void* const* p_mglo  = d_in + 40;
    void* const* p_aloc  = d_in + 46;
    void* const* p_aglo  = d_in + 52;
    const float* mtp_w   = (const float*)d_in[58];
    const float* mtp_b   = (const float*)d_in[59];
    const float* atp_w   = (const float*)d_in[60];
    const float* atp_b   = (const float*)d_in[61];

    const int nnz_l = in_sizes[0] / 2;    // 640000
    const int nnz_g = in_sizes[16] / 2;   // 2432000

    // ---- workspace carve-out ----
    const size_t SZ_G = (size_t)N_GLOBAL * DIM;
    const size_t SZ_U = (size_t)N_USERS  * DIM;
    const size_t SZ_I = (size_t)N_ITEMS  * DIM;
    float* ws = (float*)d_ws;
    float* T  = ws;            // max-sized ping buffer (also hidden buffer later)
    float* L1 = T  + SZ_G;
    float* L2 = L1 + SZ_G;
    float* GE = L2 + SZ_G;     // global_emb
    float* MC = GE + SZ_G;     // mashup_call
    float* MT = MC + SZ_U;     // mashup_tag
    float* ME = MT + SZ_U;     // mashup_emb
    float* AC = ME + SZ_U;     // api_call
    float* AT = AC + SZ_I;     // api_tag
    float* AE = AT + SZ_I;     // api_emb
    float* H  = T;             // hidden for proj_mlp (reuses T after propagates)

    // ---- output carve-out ----
    float* out = (float*)d_out;
    float* o_mf  = out;                         // mashup_final
    float* o_af  = o_mf  + SZ_U;                // api_final
    float* o_mcp = o_af  + SZ_I;                // mashup_call_proj
    float* o_mtp = o_mcp + SZ_U;                // mashup_tag_proj
    float* o_acp = o_mtp + SZ_U;                // api_call_proj
    float* o_atp = o_acp + SZ_I;                // api_tag_proj
    float* o_mlp = o_atp + SZ_I;                // mashup_local_proj
    float* o_alp = o_mlp + SZ_U;                // api_local_proj
    float* o_mgp = o_alp + SZ_I;                // mashup_global_proj
    float* o_agp = o_mgp + SZ_U;                // api_global_proj
    float* o_mtl = o_agp + SZ_I;                // mashup_tag_logits
    float* o_atl = o_mtl + (size_t)N_USERS * N_TAGS;  // api_tag_logits

    // ---- 5 propagates (3rd layer elided) ----
    run_propagate(a_mc1_i, a_mc1_v, nnz_l, a_mc2_i, a_mc2_v, nnz_l,
                  emb_mc, u_w,  N_USERS, N_ITEMS, T, L1, L2, MC, stream);
    run_propagate(a_ac1_i, a_ac1_v, nnz_l, a_ac2_i, a_ac2_v, nnz_l,
                  emb_ac, i_w,  N_ITEMS, N_USERS, T, L1, L2, AC, stream);
    run_propagate(a_mt1_i, a_mt1_v, nnz_l, a_mt2_i, a_mt2_v, nnz_l,
                  emb_mt, mt_w, N_USERS, N_TAGS,  T, L1, L2, MT, stream);
    run_propagate(a_at1_i, a_at1_v, nnz_l, a_at2_i, a_at2_v, nnz_l,
                  emb_at, at_w, N_ITEMS, N_TAGS,  T, L1, L2, AT, stream);
    run_propagate(g2_i, g2_v, nnz_g, g1_i, g1_v, nnz_g,
                  emb_g,  g_w,  N_GLOBAL, N_GLOBAL, T, L1, L2, GE, stream);

    const float* g_mashup = GE;                       // global_emb[:N_USERS]
    const float* g_api    = GE + SZ_U;                // global_emb[N_USERS:+N_ITEMS]

    // ---- view / local-global mixes ----
    combine2_kernel<<<2048, 256, 0, stream>>>(MC, MT, mvw, ME, (long)SZ_U);
    combine2_kernel<<<2048, 256, 0, stream>>>(AC, AT, avw, AE, (long)SZ_I);
    combine2_kernel<<<2048, 256, 0, stream>>>(ME, g_mashup, mlg, o_mf, (long)SZ_U);
    combine2_kernel<<<2048, 256, 0, stream>>>(AE, g_api,    alg, o_af, (long)SZ_I);

    // ---- 8 projection MLPs (WMMA GEMMs) ----
    run_proj(MC,       p_mloc, H, o_mcp, N_USERS, stream);
    run_proj(MT,       p_mloc, H, o_mtp, N_USERS, stream);
    run_proj(ME,       p_mloc, H, o_mlp, N_USERS, stream);
    run_proj(g_mashup, p_mglo, H, o_mgp, N_USERS, stream);
    run_proj(AC,       p_aloc, H, o_acp, N_ITEMS, stream);
    run_proj(AT,       p_aloc, H, o_atp, N_ITEMS, stream);
    run_proj(AE,       p_aloc, H, o_alp, N_ITEMS, stream);
    run_proj(g_api,    p_aglo, H, o_agp, N_ITEMS, stream);

    // ---- tag-prediction heads (WMMA GEMMs, N=3000) ----
    run_gemm(o_mf, mtp_w, mtp_b, o_mtl, N_USERS, N_TAGS, DIM, stream);
    run_gemm(o_af, atp_w, atp_b, o_atl, N_ITEMS, N_TAGS, DIM, stream);

    (void)n_in; (void)out_size; (void)ws_size;
}
